// RelPositionMultiHeadedAttention_72395968741693
// MI455X (gfx1250) — matrix operand order repaired and checked by reading, not compiled, not run on hardware
//
#include <hip/hip_runtime.h>
#include <hip/hip_bf16.h>

#define BB   8
#define TT   1024
#define SS   1024
#define FF   512
#define HH   8
#define DD   64
#define PPAD 2048   // padded rel-pos length (P=2047, col 2047 zeroed)

typedef unsigned short ushort_t;
typedef __attribute__((ext_vector_type(16))) __bf16 v16bf;
typedef __attribute__((ext_vector_type(8)))  __bf16 v8bf;
typedef __attribute__((ext_vector_type(8)))  float  v8f;

union ABFrag { v16bf v; v8bf h[2]; };

__device__ __forceinline__ ushort_t f32_to_bf16u(float f) {
  unsigned int u = __float_as_uint(f);
  u += 0x7FFFu + ((u >> 16) & 1u);          // round-to-nearest-even
  return (ushort_t)(u >> 16);
}

__device__ __forceinline__ v8f wmma_bf16(v16bf a, v16bf b, v8f c) {
  // D = A(16x32 bf16) * B(32x16 bf16) + C(16x16 f32)
  return __builtin_amdgcn_wmma_f32_16x16x32_bf16(false, a, false, b, (short)0, c,
                                                 false, false);
}

// A-matrix fragment (16x32, bf16), row-major source with row stride ld (elements).
// Lane layout per ISA 7.12.2: lanes 0-15 rows 0-15 / K groups {0-7,16-23};
// lanes 16-31 same rows / K groups {8-15,24-31}. Two contiguous 16B loads.
__device__ __forceinline__ v16bf load_a_frag(const ushort_t* base, int ld, int lane) {
  ABFrag f;
  const ushort_t* p = base + (size_t)(lane & 15) * ld + ((lane >> 4) << 3);
  f.h[0] = *(const v8bf*)(p);
  f.h[1] = *(const v8bf*)(p + 16);
  return f.v;
}

// B-matrix fragment (32x16, bf16), row-major KxN source with row stride ld.
// Lane l holds column N=l%16, K groups {0-7,16-23} (lanes 0-15) / {8-15,24-31} (lanes 16-31).
__device__ __forceinline__ v16bf load_b_frag(const ushort_t* base, int ld, int lane) {
  ABFrag f;
  const ushort_t* p = base + (lane & 15) + (size_t)((lane >> 4) << 3) * ld;
#pragma unroll
  for (int s = 0; s < 8; ++s) {
    f.v[s]     = __builtin_bit_cast(__bf16, p[(size_t)s * ld]);
    f.v[s + 8] = __builtin_bit_cast(__bf16, p[(size_t)(s + 16) * ld]);
  }
  return f.v;
}

// ---------------------------------------------------------------------------
// Generic projection GEMM:  C[M,512] = A[M,512] @ W[512,512] (+bias), with
// fused per-mode epilogues.
//   mode 0: Q  -> qu/qv bf16 [B,H,T,D] (+pos_bias_u/v)
//   mode 1: K  -> kT  bf16 [B,H,D,S]   + new_cache[...,0:64]  fp32
//   mode 2: V  -> v   bf16 [B,H,S,D]   + new_cache[...,64:128] fp32
//   mode 3: P  -> pT  bf16 [H*D, PPAD] (M=2047 rows, transposed)
//   mode 4: O  -> out fp32 [M,512]
// Block 256 thr (8 waves), tile 128x128, K-step 32; wave = 64x32 (8 WMMA tiles).
// ---------------------------------------------------------------------------
__global__ __launch_bounds__(256) void proj_gemm_kernel(
    const float* __restrict__ A32, const ushort_t* __restrict__ A16,
    const float* __restrict__ W, const float* __restrict__ bias,
    int M, int mode,
    void* __restrict__ out0, void* __restrict__ out1,
    const float* __restrict__ pbu, const float* __restrict__ pbv)
{
  __shared__ __align__(16) ushort_t At[128][40];
  __shared__ __align__(16) ushort_t Bt[32][136];

  const int tid  = threadIdx.x;
  const int lane = tid & 31;
  const int wave = tid >> 5;
  const int wy = wave >> 2, wx = wave & 3;         // 2x4 wave grid
  const int m0 = blockIdx.y * 128;
  const int n0 = blockIdx.x * 128;

  v8f acc[4][2];
#pragma unroll
  for (int i = 0; i < 4; ++i)
#pragma unroll
    for (int j = 0; j < 2; ++j)
      acc[i][j] = (v8f){0.f, 0.f, 0.f, 0.f, 0.f, 0.f, 0.f, 0.f};

  const int ar = tid >> 1, acb = (tid & 1) << 4;   // A stage: 128 rows x 32 cols
  const int br = tid >> 3, bcb = (tid & 7) << 4;   // B stage: 32 rows x 128 cols

  for (int k0 = 0; k0 < FF; k0 += 32) {
    __syncthreads();
    { // stage A tile (convert fp32 -> bf16, or copy bf16)
      int gr = m0 + ar;
      if (A16) {
        if (gr < M) {
          const uint4* src = (const uint4*)(A16 + (size_t)gr * FF + k0 + acb);
          *(uint4*)&At[ar][acb]     = src[0];
          *(uint4*)&At[ar][acb + 8] = src[1];
        } else {
          *(uint4*)&At[ar][acb]     = make_uint4(0u, 0u, 0u, 0u);
          *(uint4*)&At[ar][acb + 8] = make_uint4(0u, 0u, 0u, 0u);
        }
      } else {
#pragma unroll
        for (int c = 0; c < 16; c += 4) {
          float4 vsrc = (gr < M) ? *(const float4*)(A32 + (size_t)gr * FF + k0 + acb + c)
                                 : make_float4(0.f, 0.f, 0.f, 0.f);
          At[ar][acb + c + 0] = f32_to_bf16u(vsrc.x);
          At[ar][acb + c + 1] = f32_to_bf16u(vsrc.y);
          At[ar][acb + c + 2] = f32_to_bf16u(vsrc.z);
          At[ar][acb + c + 3] = f32_to_bf16u(vsrc.w);
        }
      }
    }
    { // stage B tile (W is always fp32)
#pragma unroll
      for (int c = 0; c < 16; c += 4) {
        float4 vsrc = *(const float4*)(W + (size_t)(k0 + br) * FF + n0 + bcb + c);
        Bt[br][bcb + c + 0] = f32_to_bf16u(vsrc.x);
        Bt[br][bcb + c + 1] = f32_to_bf16u(vsrc.y);
        Bt[br][bcb + c + 2] = f32_to_bf16u(vsrc.z);
        Bt[br][bcb + c + 3] = f32_to_bf16u(vsrc.w);
      }
    }
    // prefetch next k-step tiles into caches (global_prefetch_b8 on gfx1250)
    if (k0 + 32 < FF) {
      int gr = m0 + ar;
      if (A16) {
        if (gr < M) __builtin_prefetch(A16 + (size_t)gr * FF + (k0 + 32) + acb, 0, 1);
      } else {
        if (gr < M) __builtin_prefetch(A32 + (size_t)gr * FF + (k0 + 32) + acb, 0, 1);
      }
      __builtin_prefetch(W + (size_t)(k0 + 32 + br) * FF + n0 + bcb, 0, 1);
    }
    __syncthreads();

    v16bf af[4], bf[2];
#pragma unroll
    for (int i = 0; i < 4; ++i) af[i] = load_a_frag(&At[wy * 64 + i * 16][0], 40, lane);
#pragma unroll
    for (int j = 0; j < 2; ++j) bf[j] = load_b_frag(&Bt[0][wx * 32 + j * 16], 136, lane);
#pragma unroll
    for (int i = 0; i < 4; ++i)
#pragma unroll
      for (int j = 0; j < 2; ++j)
        acc[i][j] = wmma_bf16(af[i], bf[j], acc[i][j]);
  }

  // epilogue: C-layout element (row = crow+r, col = ccol) per ISA 7.12.2
  const int crow = (lane >> 4) << 3;
  const int ccol = lane & 15;
#pragma unroll
  for (int i = 0; i < 4; ++i) {
#pragma unroll
    for (int j = 0; j < 2; ++j) {
#pragma unroll
      for (int r = 0; r < 8; ++r) {
        int gm = m0 + wy * 64 + i * 16 + crow + r;
        int gn = n0 + wx * 32 + j * 16 + ccol;
        if (gm >= M) continue;
        float val = acc[i][j][r] + (bias ? bias[gn] : 0.f);
        if (mode == 0) {
          int b = gm >> 10, t = gm & (TT - 1);
          int h = gn >> 6,  d = gn & (DD - 1);
          size_t idx = ((size_t)(b * HH + h) * TT + t) * DD + d;
          ((ushort_t*)out0)[idx] = f32_to_bf16u(val + pbu[gn]);  // pbu flat [H*D]==gn
          ((ushort_t*)out1)[idx] = f32_to_bf16u(val + pbv[gn]);
        } else if (mode == 1) {
          int b = gm >> 10, s = gm & (SS - 1);
          int h = gn >> 6,  d = gn & (DD - 1);
          ((ushort_t*)out0)[((size_t)(b * HH + h) * DD + d) * SS + s] = f32_to_bf16u(val);
          ((float*)out1)[((size_t)(b * HH + h) * SS + s) * (2 * DD) + d] = val;
        } else if (mode == 2) {
          int b = gm >> 10, s = gm & (SS - 1);
          int h = gn >> 6,  d = gn & (DD - 1);
          ((ushort_t*)out0)[((size_t)(b * HH + h) * SS + s) * DD + d] = f32_to_bf16u(val);
          ((float*)out1)[((size_t)(b * HH + h) * SS + s) * (2 * DD) + DD + d] = val;
        } else if (mode == 3) {
          ((ushort_t*)out0)[(size_t)gn * PPAD + gm] = f32_to_bf16u(val);
        } else {
          ((float*)out0)[(size_t)gm * FF + gn] = val;
        }
      }
    }
  }
}

// ---------------------------------------------------------------------------
// Fused rel-pos flash attention. Grid = B*H*(T/64) blocks of 128 thr (4 waves).
// Each wave owns 16 query rows and runs the full online-softmax s-loop
// independently (no __syncthreads in the hot loop).
//   score[t,s] = (qu[t]·k[s] + qv[t]·p[T-1+s-t]) * 0.125, masked, softmax, @V
// ---------------------------------------------------------------------------
__global__ __launch_bounds__(128) void flash_attn_kernel(
    const ushort_t* __restrict__ qu, const ushort_t* __restrict__ qv,
    const ushort_t* __restrict__ kT, const ushort_t* __restrict__ vm,
    const ushort_t* __restrict__ pT, const int* __restrict__ mask,
    ushort_t* __restrict__ xctx)
{
  __shared__ __align__(16) float    g_lds[4][16][84];  // G = qv @ pT slice (16x80)
  __shared__ __align__(16) float    s_lds[4][16][68];  // combined scores (16x64)
  __shared__ __align__(16) ushort_t p_lds[4][16][72];  // exp(scores) bf16 (16x64)

  const int lane = threadIdx.x & 31;
  const int wave = threadIdx.x >> 5;
  const int blk  = blockIdx.x;
  const int tb   = blk & 15;          // t-block (T/64 = 16)
  const int bh   = blk >> 4;          // 0..63
  const int b    = bh >> 3;
  const int h    = bh & 7;
  const int t0   = tb * 64 + wave * 16;

  const ushort_t* quW = qu + ((size_t)bh * TT + t0) * DD;
  const ushort_t* qvW = qv + ((size_t)bh * TT + t0) * DD;
  const ushort_t* kTB = kT + (size_t)bh * DD * SS;
  const ushort_t* vB  = vm + (size_t)bh * SS * DD;
  const ushort_t* pTB = pT + (size_t)h * DD * PPAD;
  const int* maskB = mask + b * TT;

  v16bf quf[2], qvf[2];
  quf[0] = load_a_frag(quW,      DD, lane);
  quf[1] = load_a_frag(quW + 32, DD, lane);
  qvf[0] = load_a_frag(qvW,      DD, lane);
  qvf[1] = load_a_frag(qvW + 32, DD, lane);

  v8f o[4];
#pragma unroll
  for (int i = 0; i < 4; ++i) o[i] = (v8f){0.f,0.f,0.f,0.f,0.f,0.f,0.f,0.f};
  float m_run = -3.0e38f, l_run = 0.0f;

  const int row  = lane & 15;         // softmax lane's row
  const int ch   = lane >> 4;         // column half (32 cols each)
  const int crow = ch << 3;           // C-frag row base
  const int ccol = lane & 15;         // C-frag col

  for (int s0 = 0; s0 < SS; s0 += 64) {
    // ---- AC = qu(16x64) @ kT(64 x s-block 64) ----
    v8f ac[4];
#pragma unroll
    for (int nn = 0; nn < 4; ++nn) {
      ac[nn] = (v8f){0.f,0.f,0.f,0.f,0.f,0.f,0.f,0.f};
#pragma unroll
      for (int kk = 0; kk < 2; ++kk)
        ac[nn] = wmma_bf16(quf[kk],
            load_b_frag(kTB + (size_t)(kk * 32) * SS + s0 + nn * 16, SS, lane), ac[nn]);
    }

    // ---- G = qv(16x64) @ pT[:, pbase..pbase+79]; shift is a diagonal pick ----
    const int pbase = (TT - 1) + s0 - t0 - 15;   // always >= 0, +79 <= 2047
    v8f g[5];
#pragma unroll
    for (int nn = 0; nn < 5; ++nn) {
      g[nn] = (v8f){0.f,0.f,0.f,0.f,0.f,0.f,0.f,0.f};
#pragma unroll
      for (int kk = 0; kk < 2; ++kk)
        g[nn] = wmma_bf16(qvf[kk],
            load_b_frag(pTB + (size_t)(kk * 32) * PPAD + pbase + nn * 16, PPAD, lane), g[nn]);
    }

#pragma unroll
    for (int nn = 0; nn < 5; ++nn)
#pragma unroll
      for (int r = 0; r < 8; ++r)
        g_lds[wave][crow + r][nn * 16 + ccol] = g[nn][r];
    asm volatile("s_wait_dscnt 0x0" ::: "memory");   // same-wave LDS RAW

    // combined score: bd index = col - row + 15
#pragma unroll
    for (int nn = 0; nn < 4; ++nn)
#pragma unroll
      for (int r = 0; r < 8; ++r) {
        int rr = crow + r, cc = nn * 16 + ccol;
        s_lds[wave][rr][cc] = ac[nn][r] + g_lds[wave][rr][cc - rr + 15];
      }
    asm volatile("s_wait_dscnt 0x0" ::: "memory");

    // ---- online softmax: lane owns (row, 32-col half) ----
    int mv[32];
    {
      const int4* mp = (const int4*)(maskB + s0 + (ch << 5));   // 128B aligned
#pragma unroll
      for (int q4 = 0; q4 < 8; ++q4) {
        int4 mm = mp[q4];
        mv[q4 * 4 + 0] = mm.x; mv[q4 * 4 + 1] = mm.y;
        mv[q4 * 4 + 2] = mm.z; mv[q4 * 4 + 3] = mm.w;
      }
    }
    float sv[32];
    float tmax = -3.0e38f;
#pragma unroll
    for (int jj = 0; jj < 32; ++jj) {
      int cc = (ch << 5) + jj;
      float s = s_lds[wave][row][cc] * 0.125f;
      if (mv[jj] == 0) s = -3.0e38f;
      sv[jj] = s;
      tmax = fmaxf(tmax, s);
    }
    tmax = fmaxf(tmax, __shfl_xor(tmax, 16, 32));
    float mnew   = fmaxf(m_run, tmax);
    float cscale = __expf(m_run - mnew);
    float lloc = 0.f;
#pragma unroll
    for (int jj = 0; jj < 32; ++jj) {
      float p = __expf(sv[jj] - mnew);
      lloc += p;
      p_lds[wave][row][(ch << 5) + jj] = f32_to_bf16u(p);
    }
    lloc += __shfl_xor(lloc, 16, 32);
    l_run = l_run * cscale + lloc;
    m_run = mnew;

    // rescale O (per-row factor broadcast: lanes 0..15 hold row values)
    float rs[8];
#pragma unroll
    for (int r = 0; r < 8; ++r) rs[r] = __shfl(cscale, crow + r, 32);
#pragma unroll
    for (int nn = 0; nn < 4; ++nn)
#pragma unroll
      for (int r = 0; r < 8; ++r) o[nn][r] *= rs[r];

    asm volatile("s_wait_dscnt 0x0" ::: "memory");

    // ---- O += P(16x64) @ V(s-block 64 x 64) ----
#pragma unroll
    for (int kk = 0; kk < 2; ++kk) {
      v16bf pa = load_a_frag(&p_lds[wave][0][kk * 32], 72, lane);
#pragma unroll
      for (int nn = 0; nn < 4; ++nn)
        o[nn] = wmma_bf16(pa,
            load_b_frag(vB + (size_t)(s0 + kk * 32) * DD + nn * 16, DD, lane), o[nn]);
    }
  }

  // finalize: divide by row sums, write bf16 context [B,T,H*D]
  float linv = 1.0f / l_run;
  float rsf[8];
#pragma unroll
  for (int r = 0; r < 8; ++r) rsf[r] = __shfl(linv, crow + r, 32);
#pragma unroll
  for (int nn = 0; nn < 4; ++nn)
#pragma unroll
    for (int r = 0; r < 8; ++r) {
      int t   = t0 + crow + r;
      int col = h * DD + nn * 16 + ccol;
      xctx[(size_t)(b * TT + t) * FF + col] = f32_to_bf16u(o[nn][r] * rsf[r]);
    }
}

// ---------------------------------------------------------------------------
extern "C" void kernel_launch(void* const* d_in, const int* in_sizes, int n_in,
                              void* d_out, int out_size, void* d_ws, size_t ws_size,
                              hipStream_t stream) {
  const float* query = (const float*)d_in[0];
  const float* key   = (const float*)d_in[1];
  const float* value = (const float*)d_in[2];
  const int*   mask  = (const int*)d_in[3];
  const float* pos   = (const float*)d_in[4];
  const float* Wq = (const float*)d_in[5];  const float* bq = (const float*)d_in[6];
  const float* Wk = (const float*)d_in[7];  const float* bk = (const float*)d_in[8];
  const float* Wv = (const float*)d_in[9];  const float* bv = (const float*)d_in[10];
  const float* Wo = (const float*)d_in[11]; const float* bo = (const float*)d_in[12];
  const float* Wpos = (const float*)d_in[13];
  const float* pbu  = (const float*)d_in[14];
  const float* pbv  = (const float*)d_in[15];

  float* out   = (float*)d_out;                       // [B,T,F]
  float* cache = out + (size_t)BB * TT * FF;          // [B,H,S,2D]

  ushort_t* qu = (ushort_t*)d_ws;                     // [B,H,T,D] bf16
  ushort_t* qv = qu + (size_t)BB * HH * TT * DD;
  ushort_t* kT = qv + (size_t)BB * HH * TT * DD;      // [B,H,D,S] bf16
  ushort_t* vm = kT + (size_t)BB * HH * DD * SS;      // [B,H,S,D] bf16
  ushort_t* pT = vm + (size_t)BB * HH * SS * DD;      // [H*D, PPAD] bf16
  ushort_t* xc = pT + (size_t)HH * DD * PPAD;         // [B,T,F] bf16

  (void)hipMemsetAsync(pT, 0, (size_t)HH * DD * PPAD * sizeof(ushort_t), stream);

  dim3 blk(256);
  proj_gemm_kernel<<<dim3(4, 64), blk, 0, stream>>>(query, nullptr, Wq, bq, BB * TT, 0,
                                                    qu, qv, pbu, pbv);
  proj_gemm_kernel<<<dim3(4, 64), blk, 0, stream>>>(key, nullptr, Wk, bk, BB * SS, 1,
                                                    kT, cache, nullptr, nullptr);
  proj_gemm_kernel<<<dim3(4, 64), blk, 0, stream>>>(value, nullptr, Wv, bv, BB * SS, 2,
                                                    vm, cache, nullptr, nullptr);
  proj_gemm_kernel<<<dim3(4, 16), blk, 0, stream>>>(pos, nullptr, Wpos, nullptr,
                                                    2 * TT - 1, 3, pT, nullptr, nullptr, nullptr);

  flash_attn_kernel<<<dim3(BB * HH * (TT / 64)), dim3(128), 0, stream>>>(
      qu, qv, kT, vm, pT, mask, xc);

  proj_gemm_kernel<<<dim3(4, 64), blk, 0, stream>>>(nullptr, xc, Wo, bo, BB * TT, 4,
                                                    out, nullptr, nullptr, nullptr);
}
